// DFT_73538430042687
// MI455X (gfx1250) — matrix operand order repaired and checked by reading, not compiled, not run on hardware
//
#include <hip/hip_runtime.h>
#include <hip/hip_bf16.h>
#include <math.h>

typedef __attribute__((ext_vector_type(16))) __bf16 v16bf;
typedef __attribute__((ext_vector_type(8)))  float  v8f;
typedef int v4i_async __attribute__((vector_size(16)));   // builtin's pointee type

#define Bsz 8
#define Cc  768
#define Hh  32
#define Ww  32
#define CO  768
#define KPOS 9
#define Kdim (Cc * KPOS)     /* 6912 */
#define Nn   (Bsz * Hh * Ww) /* 8192 */

#define BM 128
#define BN 64
#define BK 32
#define KSTEPS (Kdim / BK)   /* 216 */
#define MFRAGS (CO / 16)     /* 48  */
#define NFRAGS (Nn / 16)     /* 512 */

#ifndef __has_builtin
#define __has_builtin(x) 0
#endif
#if __has_builtin(__builtin_amdgcn_global_load_async_to_lds_b128)
#define HAVE_ASYNC_LDS 1
#else
#define HAVE_ASYNC_LDS 0
#endif

#define GLOBAL_AS __attribute__((address_space(1)))
#define LDS_AS    __attribute__((address_space(3)))

// ---------------------------------------------------------------------------
// Kernel 1: offset/mask 3x3 convs (27 output channels) per spatial position.
// ---------------------------------------------------------------------------
__global__ __launch_bounds__(256) void offmask_kernel(
    const float* __restrict__ x,
    const float* __restrict__ off_w, const float* __restrict__ off_b,
    const float* __restrict__ mask_w, const float* __restrict__ mask_b,
    float* __restrict__ PY, float* __restrict__ PX, float* __restrict__ MM)
{
    __shared__ float patch[Kdim];
    __shared__ float red[256];
    const int pos = blockIdx.x;            // 0..8191
    const int b = pos >> 10;
    const int hw = pos & 1023;
    const int h = hw >> 5, w = hw & 31;
    const int tid = threadIdx.x;

    for (int j = tid; j < Kdim; j += 256) {
        const int c = j / 9, kk = j % 9;
        const int ky = kk / 3, kx = kk % 3;
        const int yy = h - 1 + ky, xx = w - 1 + kx;
        float v = 0.0f;
        if (yy >= 0 && yy < Hh && xx >= 0 && xx < Ww)
            v = x[((b * Cc + c) << 10) + (yy << 5) + xx];
        patch[j] = v;
    }
    __syncthreads();

    for (int oc = 0; oc < 27; ++oc) {
        const float* wrow = (oc < 18) ? (off_w + (size_t)oc * Kdim)
                                      : (mask_w + (size_t)(oc - 18) * Kdim);
        float p = 0.0f;
        for (int j = tid; j < Kdim; j += 256) p += patch[j] * wrow[j];
        red[tid] = p;
        __syncthreads();
        for (int s = 128; s > 0; s >>= 1) {
            if (tid < s) red[tid] += red[tid + s];
            __syncthreads();
        }
        if (tid == 0) {
            float v = red[0];
            if (oc < 18) {
                v += off_b[oc];
                v = fminf(8.0f, fmaxf(-8.0f, v));     // max_off = 32//4
                const int kk = oc >> 1;
                if ((oc & 1) == 0)
                    PY[(b * 9 + kk) * 1024 + hw] = v + (float)(h - 1) + (float)(kk / 3);
                else
                    PX[(b * 9 + kk) * 1024 + hw] = v + (float)(w - 1) + (float)(kk % 3);
            } else {
                const int kk = oc - 18;
                v += mask_b[kk];
                MM[(b * 9 + kk) * 1024 + hw] = 2.0f / (1.0f + __expf(-v));
            }
        }
        __syncthreads();
    }
}

// ---------------------------------------------------------------------------
// Fragment-layout address helpers (ISA 7.12.2, 16-bit operands, wave32):
//   A 16x32 frag: lane = m%16 + ((k%16>=8)?16:0), elem = k%8 + (k>=16?8:0)
//   B 32x16 frag: lane = n%16 + ((k%16>=8)?16:0), elem = k%8 + (k>=16?8:0)
// A fragment = 1KB: [lane][elem] => byte = lane*32 + elem*2
// ---------------------------------------------------------------------------

// Kernel 2: proj_w f32 -> bf16, pre-swizzled into A fragment layout:
//   frag index = ks*MFRAGS + m/16
__global__ __launch_bounds__(256) void aprep_kernel(
    const float* __restrict__ W, char* __restrict__ Aswz, int n)
{
    const int i = blockIdx.x * 256 + threadIdx.x;
    if (i >= n) return;
    const int m = i / Kdim, k = i % Kdim;
    const int ks = k >> 5, kk = k & 31;
    const int lane = (m & 15) + ((kk & 8) ? 16 : 0);
    const int e    = (kk & 7) + ((kk & 16) ? 8 : 0);
    const size_t addr = ((size_t)ks * MFRAGS + (m >> 4)) * 1024 + lane * 32 + e * 2;
    *(__hip_bfloat16*)(Aswz + addr) = __float2bfloat16(W[i]);
}

// Kernel 3: bilinear sampling -> P bf16 pre-swizzled into B fragment layout:
//   frag index = ks*NFRAGS + n/16
__global__ __launch_bounds__(256) void sample_kernel(
    const float* __restrict__ x,
    const float* __restrict__ PY, const float* __restrict__ PX,
    const float* __restrict__ MM,
    char* __restrict__ Pswz)
{
    const long long idx = (long long)blockIdx.x * 256 + threadIdx.x; // k*Nn + n
    const int n = (int)(idx & (Nn - 1));
    const int k = (int)(idx >> 13);           // Nn = 2^13
    const int c = k / 9, kk = k % 9;
    const int b = n >> 10, hw = n & 1023;

    const int pidx = (b * 9 + kk) * 1024 + hw;
    const float py = PY[pidx];
    const float px = PX[pidx];
    const float m  = MM[pidx];

    const float y0f = floorf(py), x0f = floorf(px);
    const int   y0 = (int)y0f,    x0 = (int)x0f;
    const float wy1 = py - y0f,   wx1 = px - x0f;

    const float* __restrict__ xp = x + ((size_t)(b * Cc + c) << 10);
    float acc = 0.0f;
#pragma unroll
    for (int dy = 0; dy < 2; ++dy) {
#pragma unroll
        for (int dx = 0; dx < 2; ++dx) {
            const int yi = y0 + dy, xi = x0 + dx;
            const float wt = (dy ? wy1 : 1.0f - wy1) * (dx ? wx1 : 1.0f - wx1);
            if (yi >= 0 && yi < Hh && xi >= 0 && xi < Ww)
                acc += wt * xp[(yi << 5) + xi];
        }
    }
    const int ks = k >> 5, kq = k & 31;
    const int lane = (n & 15) + ((kq & 8) ? 16 : 0);
    const int e    = (kq & 7) + ((kq & 16) ? 8 : 0);
    const size_t addr = ((size_t)ks * NFRAGS + (n >> 4)) * 1024 + lane * 32 + e * 2;
    *(__hip_bfloat16*)(Pswz + addr) = __float2bfloat16(acc * m);
}

// ---------------------------------------------------------------------------
// Kernel 4: GEMM  Y[768, 8192] = A[768,6912] x P[6912,8192]  via
// v_wmma_f32_16x16x32_bf16.  Operands arrive pre-swizzled, so staging is a
// contiguous global->LDS copy; use GLOBAL_LOAD_ASYNC_TO_LDS_B128 (ASYNCcnt)
// when the builtin is available, double-buffered.
// ---------------------------------------------------------------------------
__device__ __forceinline__ void wait_async_lds()
{
#if HAVE_ASYNC_LDS
#if __has_builtin(__builtin_amdgcn_s_wait_asynccnt)
    __builtin_amdgcn_s_wait_asynccnt(0);
#else
    asm volatile("s_wait_asynccnt 0x0" ::: "memory");
#endif
#endif
}

__global__ __launch_bounds__(256) void gemm_kernel(
    const char* __restrict__ Aswz,   // fragment-layout A
    const char* __restrict__ Pswz,   // fragment-layout B
    const float* __restrict__ bias,
    float* __restrict__ Y)           // NCHW f32 (d_out)
{
    __shared__ __align__(16) char As[2][8 * 1024];   // 8 M-frags
    __shared__ __align__(16) char Bs[2][4 * 1024];   // 4 N-frags

    const int tid   = threadIdx.x;
    const int lane  = tid & 31;
    const int wave  = tid >> 5;        // 0..7
    const int waveM = wave & 3;        // 32-row strip
    const int waveN = wave >> 2;       // 32-col strip
    const int m0 = blockIdx.y * BM;
    const int n0 = blockIdx.x * BN;
    const int mfrag0 = blockIdx.y * (BM / 16);  // 8 per block
    const int nfrag0 = blockIdx.x * (BN / 16);  // 4 per block

    v8f acc[2][2];
#pragma unroll
    for (int i = 0; i < 2; ++i)
#pragma unroll
        for (int j = 0; j < 2; ++j) acc[i][j] = (v8f){};

    auto stage = [&](int buf, int ks) {
        const char* asrc = Aswz + ((size_t)ks * MFRAGS + mfrag0) * 1024;
        const char* bsrc = Pswz + ((size_t)ks * NFRAGS + nfrag0) * 1024;
#if HAVE_ASYNC_LDS
        __builtin_amdgcn_global_load_async_to_lds_b128(
            (GLOBAL_AS v4i_async*)(asrc + tid * 16),
            (LDS_AS v4i_async*)(&As[buf][tid * 16]), 0, 0);
        __builtin_amdgcn_global_load_async_to_lds_b128(
            (GLOBAL_AS v4i_async*)(asrc + 4096 + tid * 16),
            (LDS_AS v4i_async*)(&As[buf][4096 + tid * 16]), 0, 0);
        __builtin_amdgcn_global_load_async_to_lds_b128(
            (GLOBAL_AS v4i_async*)(bsrc + tid * 16),
            (LDS_AS v4i_async*)(&Bs[buf][tid * 16]), 0, 0);
#else
        *(uint4*)&As[buf][tid * 16]        = *(const uint4*)(asrc + tid * 16);
        *(uint4*)&As[buf][4096 + tid * 16] = *(const uint4*)(asrc + 4096 + tid * 16);
        *(uint4*)&Bs[buf][tid * 16]        = *(const uint4*)(bsrc + tid * 16);
#endif
    };

    stage(0, 0);
    wait_async_lds();
    __syncthreads();

    for (int ks = 0; ks < KSTEPS; ++ks) {
        const int cur = ks & 1, nxt = cur ^ 1;
        if (ks + 1 < KSTEPS) stage(nxt, ks + 1);

        v16bf afr[2], bfr[2];
        afr[0] = *(const v16bf*)(As[cur] + (2 * waveM + 0) * 1024 + lane * 32);
        afr[1] = *(const v16bf*)(As[cur] + (2 * waveM + 1) * 1024 + lane * 32);
        bfr[0] = *(const v16bf*)(Bs[cur] + (2 * waveN + 0) * 1024 + lane * 32);
        bfr[1] = *(const v16bf*)(Bs[cur] + (2 * waveN + 1) * 1024 + lane * 32);
#pragma unroll
        for (int mf = 0; mf < 2; ++mf)
#pragma unroll
            for (int nf = 0; nf < 2; ++nf)
                acc[mf][nf] = __builtin_amdgcn_wmma_f32_16x16x32_bf16(
                    false, afr[mf], false, bfr[nf],
                    (short)0, acc[mf][nf], false, false);

        wait_async_lds();
        __syncthreads();
    }

    // Epilogue: C/D frag layout -> VGPR r: lanes0-15 (M=r, N=lane),
    // lanes16-31 (M=r+8, N=lane-16). Add bias, scatter to NCHW.
    const int rowadd = (lane & 16) ? 8 : 0;
    const int coll   = lane & 15;
#pragma unroll
    for (int mf = 0; mf < 2; ++mf) {
#pragma unroll
        for (int nf = 0; nf < 2; ++nf) {
#pragma unroll
            for (int r = 0; r < 8; ++r) {
                const int row = m0 + waveM * 32 + mf * 16 + r + rowadd;  // co
                const int col = n0 + waveN * 32 + nf * 16 + coll;        // n
                const int b = col >> 10, hw = col & 1023;
                Y[((b * CO + row) << 10) + hw] = acc[mf][nf][r] + bias[row];
            }
        }
    }
}

// ---------------------------------------------------------------------------
// Kernel 5: BatchNorm (batch stats) + exact GELU, in place on d_out (NCHW).
// ---------------------------------------------------------------------------
__global__ __launch_bounds__(256) void bn_gelu_kernel(
    float* __restrict__ Y, const float* __restrict__ gamma,
    const float* __restrict__ beta)
{
    __shared__ float rs[256], rs2[256];
    const int co = blockIdx.x, tid = threadIdx.x;
    float s = 0.0f, s2 = 0.0f;
    for (int i = tid; i < Nn; i += 256) {
        const int b = i >> 10, hw = i & 1023;
        const float v = Y[((b * CO + co) << 10) + hw];
        s += v;
        s2 += v * v;
    }
    rs[tid] = s; rs2[tid] = s2;
    __syncthreads();
    for (int st = 128; st > 0; st >>= 1) {
        if (tid < st) { rs[tid] += rs[tid + st]; rs2[tid] += rs2[tid + st]; }
        __syncthreads();
    }
    const float mean = rs[0] * (1.0f / Nn);
    const float var  = rs2[0] * (1.0f / Nn) - mean * mean;
    const float inv  = rsqrtf(var + 1e-5f);
    const float g = gamma[co], bt = beta[co];
    for (int i = tid; i < Nn; i += 256) {
        const int b = i >> 10, hw = i & 1023;
        const int off = ((b * CO + co) << 10) + hw;
        const float v = (Y[off] - mean) * inv * g + bt;
        Y[off] = 0.5f * v * (1.0f + erff(v * 0.70710678118654752f));
    }
}

// ---------------------------------------------------------------------------
extern "C" void kernel_launch(void* const* d_in, const int* in_sizes, int n_in,
                              void* d_out, int out_size, void* d_ws, size_t ws_size,
                              hipStream_t stream)
{
    const float* x      = (const float*)d_in[0];
    const float* proj_w = (const float*)d_in[1];
    const float* proj_b = (const float*)d_in[2];
    const float* off_w  = (const float*)d_in[3];
    const float* off_b  = (const float*)d_in[4];
    const float* mask_w = (const float*)d_in[5];
    const float* mask_b = (const float*)d_in[6];
    const float* gamma  = (const float*)d_in[7];
    const float* beta   = (const float*)d_in[8];
    float* out = (float*)d_out;

    char* ws = (char*)d_ws;
    const size_t PM_BYTES = (size_t)Bsz * 9 * 1024 * 4;      // 294912
    float* PY = (float*)(ws);
    float* PX = (float*)(ws + PM_BYTES);
    float* MM = (float*)(ws + 2 * PM_BYTES);
    char*  Aswz = ws + 3 * PM_BYTES;                          // 10.6 MB
    char*  Pswz = ws + 3 * PM_BYTES + (size_t)CO * Kdim * 2;  // 113 MB

    // 1) offsets/mask
    offmask_kernel<<<Bsz * Hh * Ww, 256, 0, stream>>>(
        x, off_w, off_b, mask_w, mask_b, PY, PX, MM);

    // 2) weight convert + swizzle to WMMA fragment layout
    const int an = CO * Kdim;
    aprep_kernel<<<(an + 255) / 256, 256, 0, stream>>>(proj_w, Aswz, an);

    // 3) bilinear sampling -> P (fragment layout)
    const long long pn = (long long)Kdim * Nn;
    sample_kernel<<<(unsigned)(pn / 256), 256, 0, stream>>>(x, PY, PX, MM, Pswz);

    // 4) WMMA GEMM -> d_out (NCHW, pre-activation)
    dim3 ggrid(Nn / BN, CO / BM);   // (128, 6)
    gemm_kernel<<<ggrid, 256, 0, stream>>>(Aswz, Pswz, proj_b, out);

    // 5) BN + GELU in place
    bn_gelu_kernel<<<CO, 256, 0, stream>>>(out, gamma, beta);
}
